// MultiHeadAttention_88905823027389
// MI455X (gfx1250) — matrix operand order repaired and checked by reading, not compiled, not run on hardware
//
#include <hip/hip_runtime.h>

// ---------------------------------------------------------------------------
// MI455X (gfx1250, wave32) GQA attention forward.
// bf16 WMMA everywhere, f32 softmax, TDM (tensor_load_to_lds) double-buffered
// GEMMs: global -> LDS via Tensor Data Mover, LDS -> VGPR via ds_load_b128,
// compute via v_wmma_f32_16x16x32_bf16.
// ---------------------------------------------------------------------------

#define BB   2
#define SS   2048
#define EE   2048
#define HH   32
#define KVH  8
#define GG   4
#define DD   64
#define KVE  (KVH * DD)   // 512

typedef __attribute__((ext_vector_type(16))) __bf16 v16bf;
typedef __attribute__((ext_vector_type(8)))  __bf16 v8bf;
typedef __attribute__((ext_vector_type(8)))  float  v8f;
typedef __attribute__((ext_vector_type(4)))  unsigned int u32x4;
typedef __attribute__((ext_vector_type(8)))  int i32x8;
typedef __attribute__((ext_vector_type(4)))  int i32x4;

static __device__ __forceinline__ v16bf combine8(v8bf lo, v8bf hi) {
  return __builtin_shufflevector(lo, hi, 0, 1, 2, 3, 4, 5, 6, 7,
                                         8, 9, 10, 11, 12, 13, 14, 15);
}

// A-matrix (16x32 bf16) load, row-major source with row stride `stride`.
// ISA layout: lanes 0-15 hold row=lane, K={0..7,16..23}; lanes 16-31 hold
// row=lane-16, K={8..15,24..31}.
static __device__ __forceinline__ v16bf load_a16x32(const __bf16* base, int stride,
                                                    int k0, int lane) {
  int r  = lane & 15;
  int kb = (lane >> 4) * 8;
  const __bf16* p = base + (size_t)r * stride + k0 + kb;
  return combine8(*(const v8bf*)p, *(const v8bf*)(p + 16));
}

// B-matrix (32x16 bf16) load from row-major (N,K) storage (i.e. W^T access):
// lane holds column n = lane&15, K = k0 + (lane>>4)*8 + {0..7,16..23}.
static __device__ __forceinline__ v16bf load_b32x16(const __bf16* base, int stride,
                                                    int n0, int k0, int lane) {
  int nn = lane & 15;
  int kb = (lane >> 4) * 8;
  const __bf16* p = base + (size_t)(n0 + nn) * stride + k0 + kb;
  return combine8(*(const v8bf*)p, *(const v8bf*)(p + 16));
}

static __device__ __forceinline__ v8f wmma_bf16(v16bf a, v16bf b, v8f c) {
  return __builtin_amdgcn_wmma_f32_16x16x32_bf16(false, a, false, b,
                                                 (short)0, c, false, false);
}

// ---------------------------------------------------------------------------
// Tensor Data Mover: 2D tile (rows x cols bf16) global -> LDS, row-major
// packed into LDS with row stride = cols.  Hand-packed D# per CDNA5 ISA
// CDNA5 ISA (group0: count/lds_addr/global_addr/type=2;
// group1: data_size=2B, tensor dims, tile dims, dim0 stride).
// This toolchain's builtin takes the 6-arg form:
//   (u32x4 g0, i32x8 g1, i32x4 g2, i32x4 g3, i32x8, i32 cpol)
// Groups 2/3 (and the trailing i32x8) are zero: 2D tensor, no iteration.
// ---------------------------------------------------------------------------
static __device__ __forceinline__ void tdm_load_2d(unsigned lds_off,
                                                   const __bf16* gsrc,
                                                   int rows, int cols,
                                                   int row_stride_elems,
                                                   int tens_cols, int tens_rows) {
  unsigned long long ga = (unsigned long long)(size_t)gsrc;
  u32x4 g0;
  g0.x = 1u;                                            // count=1, user desc
  g0.y = lds_off;                                       // lds_addr (bytes)
  g0.z = (unsigned)(ga & 0xffffffffu);                  // global_addr[31:0]
  g0.w = (unsigned)((ga >> 32) & 0x01ffffffu) | (2u << 30);  // [56:32] | type=2

  unsigned long long s0 = (unsigned long long)row_stride_elems;
  i32x8 g1;
  g1[0] = (int)(1u << 16);                              // mask=0, data_size=1 (2B)
  g1[1] = (int)(((unsigned)tens_cols & 0xffffu) << 16); // tensor_dim0[15:0]
  g1[2] = (int)((((unsigned)tens_cols >> 16) & 0xffffu) |
                (((unsigned)tens_rows & 0xffffu) << 16));      // dim0 hi | dim1 lo
  g1[3] = (int)((((unsigned)tens_rows >> 16) & 0xffffu) |
                (((unsigned)cols & 0xffffu) << 16));           // dim1 hi | tile_dim0
  g1[4] = (int)((unsigned)rows & 0xffffu);              // tile_dim1, tile_dim2=0
  g1[5] = (int)(unsigned)(s0 & 0xffffffffu);            // tensor_dim0_stride[31:0]
  g1[6] = (int)(unsigned)((s0 >> 32) & 0xffffu);        // [47:32], dim1_stride lo=0
  g1[7] = 0;
  i32x4 z4 = {0, 0, 0, 0};                              // groups 2/3 unused (2D)
  i32x8 z8 = {0, 0, 0, 0, 0, 0, 0, 0};
  __builtin_amdgcn_tensor_load_to_lds(g0, g1, z4, z4, z8, 0);
}

static __device__ __forceinline__ unsigned lds_offset_of(const void* p) {
  // addrspace(3) offset == low 32 bits of the flat LDS-aperture address
  return (unsigned)(size_t)p;
}

// ---------------------------------------------------------------------------
// f32 -> bf16 conversion
// ---------------------------------------------------------------------------
__global__ void __launch_bounds__(256)
cvt_bf16_kernel(const float* __restrict__ in, __bf16* __restrict__ out, int n) {
  int i = blockIdx.x * 256 + threadIdx.x;
  if (i < n) out[i] = (__bf16)in[i];
}

// ---------------------------------------------------------------------------
// V (b, t, kv*64+d) -> Vt (b, kv, d, t): makes P@V B-tiles contiguous in t.
// ---------------------------------------------------------------------------
__global__ void __launch_bounds__(256)
transpose_v_kernel(const __bf16* __restrict__ V, __bf16* __restrict__ Vt) {
  int i = blockIdx.x * 256 + threadIdx.x;
  if (i >= BB * SS * KVE) return;
  int d  = i & (DD - 1);
  int kv = (i >> 6) & (KVH - 1);
  int t  = (i >> 9) & (SS - 1);
  int b  = i >> 20;
  Vt[(((size_t)(b * KVH + kv)) * DD + d) * SS + t] = V[i];
}

// ---------------------------------------------------------------------------
// TDM double-buffered GEMM: Y[M,N] = Xbf[M,K] @ Wbf[N,K]^T.
// Workgroup (8 waves) owns a 128x64 macro-tile.  Per K-step of 32:
//   wave 0 issues two TENSOR_LOAD_TO_LDS (X-tile 128x32, W-tile 64x32) into
//   the ping-pong buffer, waits TENSORcnt, barrier; all waves run WMMAs from
//   the other buffer via ds_load_b128.  Each wave computes 16 rows x 64 cols.
// MODE 0: store f32.  MODE 1: store bf16 scaled by `scale`.
// ---------------------------------------------------------------------------
template <int MODE>
__global__ void __launch_bounds__(256)
gemm_tdm_kernel(const __bf16* __restrict__ A, const __bf16* __restrict__ W,
                void* __restrict__ Y, int M, int N, int K, float scale) {
  __shared__ __align__(16) __bf16 Xs[2][128 * 32];   // 2 x 8 KB
  __shared__ __align__(16) __bf16 Ws[2][64 * 32];    // 2 x 4 KB

  const int lane = threadIdx.x & 31;
  const int wid  = threadIdx.x >> 5;                 // 0..7
  const int mblocks = M >> 7;                        // M / 128
  const int mb = blockIdx.x % mblocks;
  const int nb = blockIdx.x / mblocks;

  const __bf16* Abase = A + (size_t)(mb * 128) * K;
  const __bf16* Wbase = W + (size_t)(nb * 64) * K;
  const int nk = K >> 5;

  // prologue: fill buffer 0
  if (wid == 0) {
    tdm_load_2d(lds_offset_of(&Xs[0][0]), Abase, 128, 32, K, K, M - mb * 128);
    tdm_load_2d(lds_offset_of(&Ws[0][0]), Wbase, 64, 32, K, K, N - nb * 64);
    __builtin_amdgcn_s_wait_tensorcnt(0);
  }
  __syncthreads();

  v8f acc[4] = {};
  for (int c = 0; c < nk; ++c) {
    const int cur = c & 1;
    if (wid == 0 && c + 1 < nk) {
      const int k1 = (c + 1) * 32;
      tdm_load_2d(lds_offset_of(&Xs[cur ^ 1][0]), Abase + k1, 128, 32, K,
                  K - k1, M - mb * 128);
      tdm_load_2d(lds_offset_of(&Ws[cur ^ 1][0]), Wbase + k1, 64, 32, K,
                  K - k1, N - nb * 64);
    }

    // compute from current buffer: A rows wid*16..wid*16+15, full 64 cols
    v16bf a = load_a16x32(&Xs[cur][wid * 16 * 32], 32, 0, lane);
#pragma unroll
    for (int t = 0; t < 4; ++t) {
      v16bf bm = load_b32x16(&Ws[cur][0], 32, t * 16, 0, lane);
      acc[t] = wmma_bf16(a, bm, acc[t]);
    }

    if (wid == 0 && c + 1 < nk) __builtin_amdgcn_s_wait_tensorcnt(0);
    __syncthreads();
  }

  const int n = lane & 15;
  const int g = lane >> 4;
#pragma unroll
  for (int t = 0; t < 4; ++t) {
#pragma unroll
    for (int j = 0; j < 8; ++j) {
      size_t row = (size_t)(mb * 128 + wid * 16 + j + 8 * g);
      size_t col = (size_t)(nb * 64 + t * 16 + n);
      if (MODE == 0) ((float*)Y)[row * N + col]  = acc[t][j];
      else           ((__bf16*)Y)[row * N + col] = (__bf16)(acc[t][j] * scale);
    }
  }
}

// ---------------------------------------------------------------------------
// Flash attention: one wave per (b, head, 16-query tile).  32 keys per step:
//   S(16x32) = Q(16x64) @ K^T  (4 WMMAs), causal mask + online softmax (f32),
//   P staged via LDS (C-layout out, A-layout back), O += P @ V (4 WMMAs).
// Scale 1/sqrt(D) is pre-folded into Q.
// ---------------------------------------------------------------------------
__global__ void __launch_bounds__(128)
attn_kernel(const __bf16* __restrict__ Q, const __bf16* __restrict__ Kc,
            const __bf16* __restrict__ Vt, __bf16* __restrict__ Ctx) {
  __shared__ __align__(16) __bf16 Pshared[4][16 * 32];

  const int lane = threadIdx.x & 31;
  const int wib  = threadIdx.x >> 5;
  const int w    = blockIdx.x * 4 + wib;
  const int ST   = SS / 16;                 // 128 query tiles per (b,h)
  if (w >= BB * HH * ST) return;
  const int st = w % ST;
  const int h  = (w / ST) % HH;
  const int b  = w / (ST * HH);
  const int s0 = st * 16;
  const int kv = h / GG;                    // head h = kv*G + g
  const int n  = lane & 15;
  const int g  = lane >> 4;
  __bf16* Pw = Pshared[wib];

  const __bf16* qbase = Q + ((size_t)(b * SS) + s0) * EE + h * DD;
  v16bf qa0 = load_a16x32(qbase, EE, 0, lane);
  v16bf qa1 = load_a16x32(qbase, EE, 32, lane);

  const __bf16* kbase  = Kc + (size_t)(b * SS) * KVE + kv * DD;
  const __bf16* vtbase = Vt + ((size_t)(b * KVH) + kv) * DD * SS;

  v8f   O[4] = {};
  float rowmax[8], rowsum[8];
#pragma unroll
  for (int j = 0; j < 8; ++j) { rowmax[j] = -3.0e38f; rowsum[j] = 0.0f; }

  const int nchunks = (s0 + 16 + 31) >> 5;  // causal: keys 0 .. s0+15
  for (int c = 0; c < nchunks; ++c) {
    const int t0 = c * 32;

    v16bf kb0 = load_b32x16(kbase, KVE, t0,      0,  lane);
    v16bf kb1 = load_b32x16(kbase, KVE, t0,      32, lane);
    v16bf kb2 = load_b32x16(kbase, KVE, t0 + 16, 0,  lane);
    v16bf kb3 = load_b32x16(kbase, KVE, t0 + 16, 32, lane);
    v8f Sa = {}, Sb = {};
    Sa = wmma_bf16(qa0, kb0, Sa);
    Sa = wmma_bf16(qa1, kb1, Sa);
    Sb = wmma_bf16(qa0, kb2, Sb);
    Sb = wmma_bf16(qa1, kb3, Sb);

#pragma unroll
    for (int j = 0; j < 8; ++j) {
      const int qrow = s0 + j + 8 * g;
      float sa = (qrow >= t0 + n)      ? Sa[j] : -3.0e38f;
      float sb = (qrow >= t0 + 16 + n) ? Sb[j] : -3.0e38f;

      float cm = fmaxf(sa, sb);
      cm = fmaxf(cm, __shfl_xor(cm, 1));
      cm = fmaxf(cm, __shfl_xor(cm, 2));
      cm = fmaxf(cm, __shfl_xor(cm, 4));
      cm = fmaxf(cm, __shfl_xor(cm, 8));

      float mold  = rowmax[j];
      float mn    = fmaxf(mold, cm);
      float alpha = __expf(mold - mn);
      rowmax[j]   = mn;

      float pa = __expf(sa - mn);
      float pb = __expf(sb - mn);
      float rs = pa + pb;
      rs += __shfl_xor(rs, 1);
      rs += __shfl_xor(rs, 2);
      rs += __shfl_xor(rs, 4);
      rs += __shfl_xor(rs, 8);
      rowsum[j] = rowsum[j] * alpha + rs;

      O[0][j] *= alpha; O[1][j] *= alpha; O[2][j] *= alpha; O[3][j] *= alpha;

      Pw[(j + 8 * g) * 32 + n]      = (__bf16)pa;
      Pw[(j + 8 * g) * 32 + 16 + n] = (__bf16)pb;
    }

    // O += P(16x32) @ V(32x64); same-wave LDS ops are in-order (RAW safe)
    v16bf pA;
    {
      int r  = lane & 15;
      int kb = (lane >> 4) * 8;
      const __bf16* p = Pw + r * 32 + kb;
      pA = combine8(*(const v8bf*)p, *(const v8bf*)(p + 16));
    }
#pragma unroll
    for (int db = 0; db < 4; ++db) {
      v16bf vb = load_b32x16(vtbase, SS, db * 16, t0, lane);
      O[db] = wmma_bf16(pA, vb, O[db]);
    }
  }

#pragma unroll
  for (int j = 0; j < 8; ++j) {
    float inv  = 1.0f / rowsum[j];
    size_t row = (size_t)(b * SS + s0 + j + 8 * g);
#pragma unroll
    for (int db = 0; db < 4; ++db)
      Ctx[row * EE + h * DD + db * 16 + n] = (__bf16)(O[db][j] * inv);
  }
}

// ---------------------------------------------------------------------------
// Host-side launch
// ---------------------------------------------------------------------------
extern "C" void kernel_launch(void* const* d_in, const int* in_sizes, int n_in,
                              void* d_out, int out_size, void* d_ws, size_t ws_size,
                              hipStream_t stream) {
  (void)in_sizes; (void)n_in; (void)out_size; (void)ws_size;

  const float* x  = (const float*)d_in[0];
  // d_in[1] = mask (causal tril, reconstructed analytically in-kernel)
  const float* Wq = (const float*)d_in[2];
  const float* Wk = (const float*)d_in[3];
  const float* Wv = (const float*)d_in[4];
  const float* Wo = (const float*)d_in[5];

  char* ws = (char*)d_ws;
  size_t off = 0;
  auto alloc = [&](size_t elems) -> __bf16* {
    __bf16* p = (__bf16*)(ws + off);
    off += elems * sizeof(__bf16);
    return p;
  };
  __bf16* xbf  = alloc((size_t)BB * SS * EE);   // 16 MB
  __bf16* wqbf = alloc((size_t)EE * EE);        //  8 MB
  __bf16* wkbf = alloc((size_t)KVE * EE);       //  2 MB
  __bf16* wvbf = alloc((size_t)KVE * EE);       //  2 MB
  __bf16* wobf = alloc((size_t)EE * EE);        //  8 MB
  __bf16* Qbf  = alloc((size_t)BB * SS * EE);   // 16 MB
  __bf16* Kbf  = alloc((size_t)BB * SS * KVE);  //  4 MB
  __bf16* Vbf  = alloc((size_t)BB * SS * KVE);  //  4 MB
  __bf16* Vt   = alloc((size_t)BB * SS * KVE);  //  4 MB
  __bf16* Ctx  = alloc((size_t)BB * SS * EE);   // 16 MB

  auto cvt = [&](const float* in, __bf16* out, int cnt) {
    cvt_bf16_kernel<<<(cnt + 255) / 256, 256, 0, stream>>>(in, out, cnt);
  };
  cvt(x,  xbf,  BB * SS * EE);
  cvt(Wq, wqbf, EE * EE);
  cvt(Wk, wkbf, KVE * EE);
  cvt(Wv, wvbf, KVE * EE);
  cvt(Wo, wobf, EE * EE);

  const int M = BB * SS;        // 4096 token rows
  const float qscale = 0.125f;  // 1/sqrt(HEAD_DIM=64), folded into Q

  {
    int blocks = (M / 128) * (EE / 64);   // 32 * 32
    gemm_tdm_kernel<1><<<blocks, 256, 0, stream>>>(xbf, wqbf, Qbf, M, EE, EE, qscale);
  }
  {
    int blocks = (M / 128) * (KVE / 64);  // 32 * 8
    gemm_tdm_kernel<1><<<blocks, 256, 0, stream>>>(xbf, wkbf, Kbf, M, KVE, EE, 1.0f);
    gemm_tdm_kernel<1><<<blocks, 256, 0, stream>>>(xbf, wvbf, Vbf, M, KVE, EE, 1.0f);
  }

  transpose_v_kernel<<<(BB * SS * KVE + 255) / 256, 256, 0, stream>>>(Vbf, Vt);

  {
    int total_waves = BB * HH * (SS / 16);  // 8192
    attn_kernel<<<total_waves / 4, 128, 0, stream>>>(Qbf, Kbf, Vt, Ctx);
  }

  {
    int blocks = (M / 128) * (EE / 64);
    gemm_tdm_kernel<0><<<blocks, 256, 0, stream>>>(Ctx, wobf, d_out, M, EE, EE, 1.0f);
  }
}
